// MultiHeadAttention_34729105555669
// MI455X (gfx1250) — matrix operand order repaired and checked by reading, not compiled, not run on hardware
//
#include <hip/hip_runtime.h>

typedef __attribute__((ext_vector_type(16))) _Float16 v16h;
typedef __attribute__((ext_vector_type(8)))  _Float16 v8h;
typedef __attribute__((ext_vector_type(4)))  _Float16 v4h;
typedef __attribute__((ext_vector_type(8)))  float    v8f;

#define B_S 4
#define SEQ 2048
#define DM  512
#define NH  8
#define DK  64

// ---------------- WMMA fragment helpers (CDNA5 16x16x32 f16 layouts) ----------------

// A matrix 16x32 (MxK), row-major source, ld = row stride in elements.
// lanes 0-15: row=lane, K = {0..7} and {16..23}; lanes 16-31: row=lane-16, K = {8..15} and {24..31}
static __device__ __forceinline__ v16h load_matA(const _Float16* __restrict__ p0, int ld) {
  const int lane = threadIdx.x & 31;
  const _Float16* p = p0 + (size_t)(lane & 15) * ld + ((lane >> 4) << 3);
  v8h lo = *(const v8h*)(p);
  v8h hi = *(const v8h*)(p + 16);
  return __builtin_shufflevector(lo, hi, 0,1,2,3,4,5,6,7,8,9,10,11,12,13,14,15);
}

// B matrix 32x16 (KxN). Source stored so each N column's K values are contiguous:
// element (k,n) at p0[n*ld + k].  lanes 0-15: n=lane, K = {0..7} and {16..23}; lanes 16-31: n=lane-16, K = {8..15} and {24..31}
static __device__ __forceinline__ v16h load_matB(const _Float16* __restrict__ p0, int ld) {
  const int lane = threadIdx.x & 31;
  const _Float16* p = p0 + (size_t)(lane & 15) * ld + ((lane >> 4) << 3);
  v8h lo = *(const v8h*)(p);
  v8h hi = *(const v8h*)(p + 16);
  return __builtin_shufflevector(lo, hi, 0,1,2,3,4,5,6,7,8,9,10,11,12,13,14,15);
}

static __device__ __forceinline__ v8f wmma_f16(v16h a, v16h b, v8f c) {
  return __builtin_amdgcn_wmma_f32_16x16x32_f16(false, a, false, b, (short)0, c, false, false);
}

static __device__ __forceinline__ float rowmax16(float v) {
#pragma unroll
  for (int off = 1; off < 16; off <<= 1) v = fmaxf(v, __shfl_xor(v, off, 32));
  return v;
}
static __device__ __forceinline__ float rowsum16(float v) {
#pragma unroll
  for (int off = 1; off < 16; off <<= 1) v += __shfl_xor(v, off, 32);
  return v;
}

// ---------------- kernels ----------------

__global__ void f32_to_f16_kernel(const float* __restrict__ src, _Float16* __restrict__ dst, int n) {
  int i = (blockIdx.x * blockDim.x + threadIdx.x) * 4;
  if (i < n) {
    float4 f = *(const float4*)(src + i);
    v4h h = {(_Float16)f.x, (_Float16)f.y, (_Float16)f.z, (_Float16)f.w};
    *(v4h*)(dst + i) = h;
  }
}

// out[m,n] = sum_k A[m,k] * W[n,k] + bias[n]   (A: [M,512] f16, W: [512,512] f16, N=512)
// W strip (64 cols x 512 k = 64 KB) staged in LDS via async global->LDS DMA, shared by 8 waves.
// Per wave: 32x64 output tile, K fully unrolled.
// mode 0: f16 row-major [M,512]
// mode 1: f16 head-transposed for V: [b, h, d, l]
// mode 2: f32 row-major [M,512]
__global__ __launch_bounds__(256) void gemm_wmma_kernel(
    const _Float16* __restrict__ A, const _Float16* __restrict__ W,
    const float* __restrict__ bias, void* __restrict__ out, int M, int mode)
{
  __shared__ _Float16 wtile[64 * DM];   // 64 KB

  const int K = DM;
  const int lane = threadIdx.x & 31;
  const int wave = threadIdx.x >> 5;
  const int mbase = blockIdx.x * 256 + wave * 32;
  const int nbase = blockIdx.y * 64;

  // async stage of W strip: rows nbase..nbase+63, all K (64 KB), no VGPR round-trip.
  // inst offset applies to BOTH global and LDS addresses (ISA 08_async_tensor §4.4).
  {
    const _Float16* wsrc = W + (size_t)nbase * K;
    unsigned laddr = (unsigned)(size_t)(&wtile[0]) + threadIdx.x * 16u;
    unsigned long long gaddr = (unsigned long long)(size_t)wsrc + threadIdx.x * 16u;
    asm volatile("global_load_async_to_lds_b128 %0, %1, off offset:0"     :: "v"(laddr), "v"(gaddr) : "memory");
    asm volatile("global_load_async_to_lds_b128 %0, %1, off offset:4096"  :: "v"(laddr), "v"(gaddr) : "memory");
    asm volatile("global_load_async_to_lds_b128 %0, %1, off offset:8192"  :: "v"(laddr), "v"(gaddr) : "memory");
    asm volatile("global_load_async_to_lds_b128 %0, %1, off offset:12288" :: "v"(laddr), "v"(gaddr) : "memory");
    asm volatile("global_load_async_to_lds_b128 %0, %1, off offset:16384" :: "v"(laddr), "v"(gaddr) : "memory");
    asm volatile("global_load_async_to_lds_b128 %0, %1, off offset:20480" :: "v"(laddr), "v"(gaddr) : "memory");
    asm volatile("global_load_async_to_lds_b128 %0, %1, off offset:24576" :: "v"(laddr), "v"(gaddr) : "memory");
    asm volatile("global_load_async_to_lds_b128 %0, %1, off offset:28672" :: "v"(laddr), "v"(gaddr) : "memory");
    asm volatile("global_load_async_to_lds_b128 %0, %1, off offset:32768" :: "v"(laddr), "v"(gaddr) : "memory");
    asm volatile("global_load_async_to_lds_b128 %0, %1, off offset:36864" :: "v"(laddr), "v"(gaddr) : "memory");
    asm volatile("global_load_async_to_lds_b128 %0, %1, off offset:40960" :: "v"(laddr), "v"(gaddr) : "memory");
    asm volatile("global_load_async_to_lds_b128 %0, %1, off offset:45056" :: "v"(laddr), "v"(gaddr) : "memory");
    asm volatile("global_load_async_to_lds_b128 %0, %1, off offset:49152" :: "v"(laddr), "v"(gaddr) : "memory");
    asm volatile("global_load_async_to_lds_b128 %0, %1, off offset:53248" :: "v"(laddr), "v"(gaddr) : "memory");
    asm volatile("global_load_async_to_lds_b128 %0, %1, off offset:57344" :: "v"(laddr), "v"(gaddr) : "memory");
    asm volatile("global_load_async_to_lds_b128 %0, %1, off offset:61440" :: "v"(laddr), "v"(gaddr) : "memory");
    asm volatile("s_wait_asynccnt 0x0" ::: "memory");
  }
  __syncthreads();

  const v8f vzero = {0,0,0,0,0,0,0,0};
  v8f acc[2][4] = {{vzero,vzero,vzero,vzero},{vzero,vzero,vzero,vzero}};

  const _Float16* a0p = A + (size_t)mbase * K;
  const _Float16* a1p = a0p + (size_t)16 * K;
  const int bk_off = ((lane >> 4) << 3);
  const int bn = (lane & 15);

#pragma unroll
  for (int kb = 0; kb < K; kb += 32) {
    v16h a0 = load_matA(a0p + kb, K);
    v16h a1 = load_matA(a1p + kb, K);
#pragma unroll
    for (int t = 0; t < 4; t++) {
      v8h blo = *(const v8h*)&wtile[(t * 16 + bn) * K + kb + bk_off];
      v8h bhi = *(const v8h*)&wtile[(t * 16 + bn) * K + kb + bk_off + 16];
      v16h b = __builtin_shufflevector(blo, bhi, 0,1,2,3,4,5,6,7,8,9,10,11,12,13,14,15);
      acc[0][t] = wmma_f16(a0, b, acc[0][t]);
      acc[1][t] = wmma_f16(a1, b, acc[1][t]);
    }
  }

  const int col  = lane & 15;
  const int rtop = (lane >> 4) << 3;
#pragma unroll
  for (int u = 0; u < 2; u++) {
#pragma unroll
    for (int t = 0; t < 4; t++) {
      const int n = nbase + t * 16 + col;
      const float bv = bias[n];
#pragma unroll
      for (int r = 0; r < 8; r++) {
        const int row = mbase + u * 16 + rtop + r;
        const float v = acc[u][t][r] + bv;
        if (mode == 0) {
          ((_Float16*)out)[(size_t)row * DM + n] = (_Float16)v;
        } else if (mode == 1) {
          const int bb = row >> 11, l = row & (SEQ - 1);
          const int h = n >> 6, d = n & 63;
          ((_Float16*)out)[((size_t)((bb * NH + h) * DK + d) << 11) + l] = (_Float16)v;
        } else {
          ((float*)out)[(size_t)row * DM + n] = v;
        }
      }
    }
  }
}

// Causal flash attention. Q,K: [b, l, h*64+d] f16 (row-major). V: [b, h, d, l] f16.
// Z out: [b, l, h*64+d] f16.  One wave per (b, h, 16-row q tile); 64-column k-steps.
__global__ __launch_bounds__(256) void attn_wmma_kernel(
    const _Float16* __restrict__ Q, const _Float16* __restrict__ Kc,
    const _Float16* __restrict__ Vt, _Float16* __restrict__ Z)
{
  __shared__ _Float16 pst[8][16 * 64];   // per-wave 16x64 P staging (C-layout -> A-layout)

  const int lane = threadIdx.x & 31;
  const int wave = threadIdx.x >> 5;
  const int task = blockIdx.x * 8 + wave;        // 4096 tasks
  const int qt = task & 127;
  const int h  = (task >> 7) & 7;
  const int b  = task >> 10;
  const int qbase = qt << 4;

  const size_t bh = (size_t)b * SEQ * DM + (size_t)h * DK;
  const _Float16* qp = Q + bh + (size_t)qbase * DM;
  const v16h aq0 = load_matA(qp, DM);        // d = 0..31
  const v16h aq1 = load_matA(qp + 32, DM);   // d = 32..63

  const v8f vzero = {0,0,0,0,0,0,0,0};
  v8f o0 = vzero, o1 = vzero, o2 = vzero, o3 = vzero;
  float mr[8], lr[8];
#pragma unroll
  for (int r = 0; r < 8; r++) { mr[r] = -1.0e30f; lr[r] = 0.0f; }

  const int col  = lane & 15;
  const int rtop = (lane >> 4) << 3;
  const int prow = lane & 15;
  const int pk0  = (lane >> 4) << 3;
  const _Float16* vb = Vt + (size_t)(b * NH + h) * DK * SEQ;
  const float SC = 0.180336880111120419f;    // (1/8) * log2(e)

  for (int kb = 0; kb < qbase + 16; kb += 64) {
    // ---- S(16x64) = Q(16x64) @ K^T(64x64): 4 C tiles, 2 WMMAs each ----
    const _Float16* kp = Kc + bh + (size_t)kb * DM;
    v8f s0 = vzero, s1 = vzero, s2 = vzero, s3 = vzero;
    {
      v16h bka, bkb;
      bka = load_matB(kp, DM);                       bkb = load_matB(kp + 32, DM);
      s0 = wmma_f16(aq0, bka, s0);                   s0 = wmma_f16(aq1, bkb, s0);
      bka = load_matB(kp + (size_t)16 * DM, DM);     bkb = load_matB(kp + (size_t)16 * DM + 32, DM);
      s1 = wmma_f16(aq0, bka, s1);                   s1 = wmma_f16(aq1, bkb, s1);
      bka = load_matB(kp + (size_t)32 * DM, DM);     bkb = load_matB(kp + (size_t)32 * DM + 32, DM);
      s2 = wmma_f16(aq0, bka, s2);                   s2 = wmma_f16(aq1, bkb, s2);
      bka = load_matB(kp + (size_t)48 * DM, DM);     bkb = load_matB(kp + (size_t)48 * DM + 32, DM);
      s3 = wmma_f16(aq0, bka, s3);                   s3 = wmma_f16(aq1, bkb, s3);
    }

    // issue V low-K fragments now; softmax below hides their latency
    v16h bvl0 = load_matB(vb + kb, SEQ);
    v16h bvl1 = load_matB(vb + (size_t)16 * SEQ + kb, SEQ);
    v16h bvl2 = load_matB(vb + (size_t)32 * SEQ + kb, SEQ);
    v16h bvl3 = load_matB(vb + (size_t)48 * SEQ + kb, SEQ);

    // ---- causal mask + online softmax (exp2 domain) ----
    const int c0 = kb + col;
#pragma unroll
    for (int r = 0; r < 8; r++) {
      const int row = qbase + rtop + r;
      float v0 = (c0      <= row) ? s0[r] * SC : -1.0e30f;
      float v1 = (c0 + 16 <= row) ? s1[r] * SC : -1.0e30f;
      float v2 = (c0 + 32 <= row) ? s2[r] * SC : -1.0e30f;
      float v3 = (c0 + 48 <= row) ? s3[r] * SC : -1.0e30f;
      float tm = rowmax16(fmaxf(fmaxf(v0, v1), fmaxf(v2, v3)));
      float mn = fmaxf(mr[r], tm);
      float corr = __builtin_amdgcn_exp2f(mr[r] - mn);
      float p0 = __builtin_amdgcn_exp2f(v0 - mn);
      float p1 = __builtin_amdgcn_exp2f(v1 - mn);
      float p2 = __builtin_amdgcn_exp2f(v2 - mn);
      float p3 = __builtin_amdgcn_exp2f(v3 - mn);
      lr[r] = lr[r] * corr + rowsum16((p0 + p1) + (p2 + p3));
      mr[r] = mn;
      o0[r] *= corr; o1[r] *= corr; o2[r] *= corr; o3[r] *= corr;
      const int pr = (rtop + r) * 64 + col;
      pst[wave][pr]      = (_Float16)p0;
      pst[wave][pr + 16] = (_Float16)p1;
      pst[wave][pr + 32] = (_Float16)p2;
      pst[wave][pr + 48] = (_Float16)p3;
    }
    asm volatile("s_wait_dscnt 0x0" ::: "memory");  // wave-local LDS store->load fence

    // ---- O(16x64) += P(16x64) @ V(64x64) ----
    v8h plo = *(const v8h*)&pst[wave][prow * 64 + pk0];
    v8h phi = *(const v8h*)&pst[wave][prow * 64 + pk0 + 16];
    v16h pa0 = __builtin_shufflevector(plo, phi, 0,1,2,3,4,5,6,7,8,9,10,11,12,13,14,15);
    plo = *(const v8h*)&pst[wave][prow * 64 + pk0 + 32];
    phi = *(const v8h*)&pst[wave][prow * 64 + pk0 + 48];
    v16h pa1 = __builtin_shufflevector(plo, phi, 0,1,2,3,4,5,6,7,8,9,10,11,12,13,14,15);

    o0 = wmma_f16(pa0, bvl0, o0);
    o1 = wmma_f16(pa0, bvl1, o1);
    o2 = wmma_f16(pa0, bvl2, o2);
    o3 = wmma_f16(pa0, bvl3, o3);
    v16h bvh0 = load_matB(vb + kb + 32, SEQ);
    v16h bvh1 = load_matB(vb + (size_t)16 * SEQ + kb + 32, SEQ);
    v16h bvh2 = load_matB(vb + (size_t)32 * SEQ + kb + 32, SEQ);
    v16h bvh3 = load_matB(vb + (size_t)48 * SEQ + kb + 32, SEQ);
    o0 = wmma_f16(pa1, bvh0, o0);
    o1 = wmma_f16(pa1, bvh1, o1);
    o2 = wmma_f16(pa1, bvh2, o2);
    o3 = wmma_f16(pa1, bvh3, o3);
  }

  // ---- normalize and store Z ----
#pragma unroll
  for (int r = 0; r < 8; r++) {
    const float inv = 1.0f / lr[r];
    const int row = qbase + rtop + r;
    _Float16* zp = Z + (size_t)(b * SEQ + row) * DM + h * DK;
    zp[col]      = (_Float16)(o0[r] * inv);
    zp[col + 16] = (_Float16)(o1[r] * inv);
    zp[col + 32] = (_Float16)(o2[r] * inv);
    zp[col + 48] = (_Float16)(o3[r] * inv);
  }
}

// ---------------- host ----------------

extern "C" void kernel_launch(void* const* d_in, const int* in_sizes, int n_in,
                              void* d_out, int out_size, void* d_ws, size_t ws_size,
                              hipStream_t stream)
{
  (void)in_sizes; (void)n_in; (void)out_size; (void)ws_size;

  const float* x  = (const float*)d_in[0];
  const float* wq = (const float*)d_in[1];
  const float* bq = (const float*)d_in[2];
  const float* wk = (const float*)d_in[3];
  const float* bk = (const float*)d_in[4];
  const float* wv = (const float*)d_in[5];
  const float* bv = (const float*)d_in[6];
  const float* wo = (const float*)d_in[7];
  const float* bo = (const float*)d_in[8];

  const int M = B_S * SEQ;              // 8192
  const size_t NX = (size_t)M * DM;     // 4,194,304
  const size_t NW = (size_t)DM * DM;    // 262,144

  _Float16* ws   = (_Float16*)d_ws;
  _Float16* x16  = ws;                  // [M,512]
  _Float16* q16  = x16  + NX;           // [b,l,h*64+d]
  _Float16* k16  = q16  + NX;           // [b,l,h*64+d]
  _Float16* vT16 = k16  + NX;           // [b,h,d,l]
  _Float16* wq16 = vT16 + NX;
  _Float16* wk16 = wq16 + NW;
  _Float16* wv16 = wk16 + NW;
  _Float16* wo16 = wv16 + NW;
  _Float16* z16  = x16;                 // reuse x16 after it's consumed

  const int CT = 256;
  f32_to_f16_kernel<<<(int)((NX / 4 + CT - 1) / CT), CT, 0, stream>>>(x,  x16,  (int)NX);
  f32_to_f16_kernel<<<(int)((NW / 4 + CT - 1) / CT), CT, 0, stream>>>(wq, wq16, (int)NW);
  f32_to_f16_kernel<<<(int)((NW / 4 + CT - 1) / CT), CT, 0, stream>>>(wk, wk16, (int)NW);
  f32_to_f16_kernel<<<(int)((NW / 4 + CT - 1) / CT), CT, 0, stream>>>(wv, wv16, (int)NW);
  f32_to_f16_kernel<<<(int)((NW / 4 + CT - 1) / CT), CT, 0, stream>>>(wo, wo16, (int)NW);

  dim3 ggrid(M / 256, DM / 64);
  gemm_wmma_kernel<<<ggrid, 256, 0, stream>>>(x16, wq16, bq, q16,  M, 0);
  gemm_wmma_kernel<<<ggrid, 256, 0, stream>>>(x16, wk16, bk, k16,  M, 0);
  gemm_wmma_kernel<<<ggrid, 256, 0, stream>>>(x16, wv16, bv, vT16, M, 1);

  attn_wmma_kernel<<<(B_S * NH * (SEQ / 16)) / 8, 256, 0, stream>>>(q16, k16, vT16, z16);

  gemm_wmma_kernel<<<ggrid, 256, 0, stream>>>(z16, wo16, bo, d_out, M, 2);
}
